// TestLSTM_29137058136622
// MI455X (gfx1250) — matrix operand order repaired and checked by reading, not compiled, not run on hardware
//
#include <hip/hip_runtime.h>
#include <math.h>

// Problem constants (from reference)
#define T_STEPS 65536
#define I_DIM   512
#define H_DIM   512
#define G_DIM   2048   // 4*H

typedef _Float16 half_t;
typedef __attribute__((ext_vector_type(16))) _Float16 v16h;
typedef __attribute__((ext_vector_type(8)))  _Float16 h8;
typedef __attribute__((ext_vector_type(8)))  float    v8f;
typedef int v4i __attribute__((vector_size(16)));   // matches builtin param type

#define GLOBAL_AS __attribute__((address_space(1)))
#define LDS_AS    __attribute__((address_space(3)))

#if __has_builtin(__builtin_amdgcn_global_load_async_to_lds_b128) && \
    __has_builtin(__builtin_amdgcn_s_wait_asynccnt)
#define HAVE_ASYNC_LDS 1
#else
#define HAVE_ASYNC_LDS 0
#endif

// ---------------------------------------------------------------------------
// fp32 -> f16 conversion (grid-stride)
// ---------------------------------------------------------------------------
__global__ void cvt_f32_to_f16(const float* __restrict__ in,
                               half_t* __restrict__ out, int n) {
  int i = blockIdx.x * blockDim.x + threadIdx.x;
  int stride = gridDim.x * blockDim.x;
  for (; i < n; i += stride) out[i] = (half_t)in[i];
}

__global__ void init_state(float* hbuf, int* bar) {
  for (int i = threadIdx.x; i < 2 * H_DIM; i += blockDim.x) hbuf[i] = 0.f;
  if (threadIdx.x == 0) *bar = 0;
}

// ---------------------------------------------------------------------------
// x_proj GEMM:  Xp[t][n] = sum_k A[t][k] * W[n][k] + b_ih[n]   (f16 in, f32 acc)
// Block tile 128x128, 8 waves (2x4), each wave 64x32 via 4x2 WMMA 16x16x32 f16.
// LDS double-buffered; tiles fetched with GLOBAL_LOAD_ASYNC_TO_LDS_B128
// (ASYNCcnt) so the copy of slab k+1 overlaps the WMMAs of slab k.
// ---------------------------------------------------------------------------
#define BM  128
#define BN  128
#define KC  32
#define LDT 40   // padded LDS stride (halfs): rows stay 16B-aligned

__global__ __launch_bounds__(256) void gemm_xproj(
    const half_t* __restrict__ A,    // [T][512]
    const half_t* __restrict__ W,    // [2048][512]  (row n, col k)
    const float*  __restrict__ bih,  // [2048]
    half_t* __restrict__ Xp)         // [T][2048]
{
  __shared__ half_t As[2][BM * LDT];
  __shared__ half_t Bs[2][BN * LDT];

  const int tid  = threadIdx.x;
  const int lane = tid & 31;
  const int wave = tid >> 5;
  const int bm = blockIdx.x * BM;
  const int bn = blockIdx.y * BN;
  const int wm = (wave >> 2) * 64;   // 0 or 64
  const int wn = (wave & 3) * 32;    // 0,32,64,96
  const int l16 = lane & 15;
  const int lhi = lane >> 4;         // 0 or 1

  v8f acc[4][2] = {};

  // stage one 128x32 A-tile + 128x32 W-tile into LDS buffer `buf`
  auto stage = [&](int buf, int k0) {
    #pragma unroll
    for (int s = tid; s < BM * 4; s += 256) {
      int r = s >> 2, seg = (s & 3) * 8;
      const half_t* ga = A + (size_t)(bm + r) * I_DIM + k0 + seg;
      const half_t* gb = W + (size_t)(bn + r) * I_DIM + k0 + seg;
#if HAVE_ASYNC_LDS
      __builtin_amdgcn_global_load_async_to_lds_b128(
          (GLOBAL_AS v4i*)ga,
          (LDS_AS v4i*)&As[buf][r * LDT + seg], 0, 0);
      __builtin_amdgcn_global_load_async_to_lds_b128(
          (GLOBAL_AS v4i*)gb,
          (LDS_AS v4i*)&Bs[buf][r * LDT + seg], 0, 0);
#else
      *(h8*)&As[buf][r * LDT + seg] = *(const h8*)ga;
      *(h8*)&Bs[buf][r * LDT + seg] = *(const h8*)gb;
#endif
    }
  };

  int cur = 0;
  stage(cur, 0);
#if HAVE_ASYNC_LDS
  __builtin_amdgcn_s_wait_asynccnt(0);
#endif
  __syncthreads();

  for (int k0 = 0; k0 < I_DIM; k0 += KC) {
    const int nxt = cur ^ 1;
    if (k0 + KC < I_DIM) stage(nxt, k0 + KC);   // async prefetch of next slab

    // ---- build fragments per ISA 7.12.2 layouts, consume buffer `cur` ----
    v16h afr[4], bfr[2];
    #pragma unroll
    for (int mt = 0; mt < 4; ++mt) {
      // lane<16: M=l16, K={0..7,16..23}; lane>=16: M=l16, K={8..15,24..31}
      const half_t* p = &As[cur][(wm + mt * 16 + l16) * LDT + lhi * 8];
      h8 lo = *(const h8*)p;
      h8 hi = *(const h8*)(p + 16);
      afr[mt] = __builtin_shufflevector(lo, hi, 0,1,2,3,4,5,6,7,8,9,10,11,12,13,14,15);
    }
    #pragma unroll
    for (int nt = 0; nt < 2; ++nt) {
      // B mirrored over N: lane<16: N=l16, K={0..7,16..23}; lane>=16: N=l16, K={8..15,24..31}
      const half_t* p = &Bs[cur][(wn + nt * 16 + l16) * LDT + lhi * 8];
      h8 lo = *(const h8*)p;
      h8 hi = *(const h8*)(p + 16);
      bfr[nt] = __builtin_shufflevector(lo, hi, 0,1,2,3,4,5,6,7,8,9,10,11,12,13,14,15);
    }

    #pragma unroll
    for (int mt = 0; mt < 4; ++mt)
      #pragma unroll
      for (int nt = 0; nt < 2; ++nt)
        acc[mt][nt] = __builtin_amdgcn_wmma_f32_16x16x32_f16(
            false, afr[mt], false, bfr[nt], (short)0, acc[mt][nt], false, false);

#if HAVE_ASYNC_LDS
    __builtin_amdgcn_s_wait_asynccnt(0);   // prefetch of `nxt` landed
#endif
    __syncthreads();                        // frag reads of `cur` done everywhere
    cur = nxt;
  }

  // ---- epilogue: +b_ih, f16 store. D layout: M = (lane>=16)*8+v, N = lane%16
  #pragma unroll
  for (int mt = 0; mt < 4; ++mt) {
    #pragma unroll
    for (int nt = 0; nt < 2; ++nt) {
      int ncol = bn + wn + nt * 16 + l16;
      float bv = bih[ncol];
      #pragma unroll
      for (int v = 0; v < 8; ++v) {
        int mrow = bm + wm + mt * 16 + lhi * 8 + v;
        Xp[(size_t)mrow * G_DIM + ncol] = (half_t)(acc[mt][nt][v] + bv);
      }
    }
  }
}

// ---------------------------------------------------------------------------
// Persistent LSTM recurrence: 16 WGs, each owns 32 h-elements => 128 W_hh rows
// (4 gates x 32) pinned in 256 KB of LDS (CDNA5: 320 KB/WGP). Per step:
//   - stage h (2 KB) from L2 with agent-scope loads
//   - 128x512 matvec (2 threads/row, float4 LDS reads, shfl-pair reduce)
//   - gate nonlinearities on 32 lanes, publish h slice, global counter barrier
// ---------------------------------------------------------------------------
#define RBLK 16
#define RTHR 256

__device__ __forceinline__ float sigmf(float x) { return 1.f / (1.f + __expf(-x)); }

__global__ __launch_bounds__(RTHR) void lstm_rec(
    const half_t* __restrict__ Xp,   // [T][2048]
    const float* __restrict__ Whh,   // [2048][512]
    const float* __restrict__ bhh,   // [2048]
    float* hbuf,                     // 2 x [512], double-buffered
    int* bar)
{
  extern __shared__ float smem[];
  float* Ws    = smem;               // [128][512]
  float* gates = Ws + 128 * 512;     // [128]
  float* biasr = gates + 128;        // [128]
  float* c_s   = biasr + 128;        // [32]
  float* h_s   = c_s + 32;           // [512]

  const int b = blockIdx.x, tid = threadIdx.x;

  // local row r = gate*32 + j  ->  global W_hh row = gate*512 + b*32 + j
  for (int idx = tid; idx < 128 * 512; idx += RTHR) {
    int r = idx >> 9, k = idx & 511;
    int grow = ((r >> 5) << 9) | (b << 5) | (r & 31);
    Ws[idx] = Whh[(size_t)grow * 512 + k];
  }
  if (tid < 128) {
    int grow = ((tid >> 5) << 9) | (b << 5) | (tid & 31);
    biasr[tid] = bhh[grow];
  }
  if (tid < 32) c_s[tid] = 0.f;
  __syncthreads();

  const int row   = tid >> 1;           // 0..127
  const int halfk = (tid & 1) * 256;    // K split between lane pairs
  const float4* w4 = (const float4*)(Ws + row * 512 + halfk);
  const int xrow = ((row >> 5) << 9) | (b << 5) | (row & 31);

  for (int t = 0; t < T_STEPS; ++t) {
    // stage h (device-scope so other WGPs' stores are seen)
    const float* hc = hbuf + (t & 1) * H_DIM;
    for (int i = tid; i < H_DIM; i += RTHR)
      h_s[i] = __hip_atomic_load(hc + i, __ATOMIC_RELAXED, __HIP_MEMORY_SCOPE_AGENT);
    __syncthreads();

    float acc = 0.f;
    const float4* h4 = (const float4*)(h_s + halfk);
    #pragma unroll 8
    for (int k = 0; k < 64; ++k) {
      float4 w = w4[k], h = h4[k];
      acc = __builtin_fmaf(w.x, h.x, acc);
      acc = __builtin_fmaf(w.y, h.y, acc);
      acc = __builtin_fmaf(w.z, h.z, acc);
      acc = __builtin_fmaf(w.w, h.w, acc);
    }
    acc += __shfl_xor(acc, 1, 32);      // pair (2r, 2r+1) -> full row sum
    if ((tid & 1) == 0)
      gates[row] = acc + biasr[row] + (float)Xp[(size_t)t * G_DIM + xrow];
    __syncthreads();

    if (tid < 32) {
      float ig = sigmf(gates[tid]);
      float fg = sigmf(gates[32 + tid]);
      float gg = tanhf(gates[64 + tid]);
      float og = sigmf(gates[96 + tid]);
      float c  = fg * c_s[tid] + ig * gg;
      c_s[tid] = c;
      float h  = og * tanhf(c);
      __hip_atomic_store(hbuf + ((t + 1) & 1) * H_DIM + (b << 5) + tid, h,
                         __ATOMIC_RELEASE, __HIP_MEMORY_SCOPE_AGENT);
    }
    __syncthreads();

    // inter-WGP barrier: monotonic counter, one spinner per block, s_sleep poll
    if (tid == 0) {
      __hip_atomic_fetch_add(bar, 1, __ATOMIC_ACQ_REL, __HIP_MEMORY_SCOPE_AGENT);
      const int target = RBLK * (t + 1);
      while (__hip_atomic_load(bar, __ATOMIC_ACQUIRE, __HIP_MEMORY_SCOPE_AGENT) < target)
        __builtin_amdgcn_s_sleep(1);
    }
    __syncthreads();
  }
}

// ---------------------------------------------------------------------------
// softmax over the final h (512 elems), one block of 256 threads (8 waves)
// ---------------------------------------------------------------------------
__global__ __launch_bounds__(256) void softmax512(const float* __restrict__ hin,
                                                  float* __restrict__ out) {
  __shared__ float red[8];
  __shared__ float smax, ssum;
  int tid = threadIdx.x;
  float v0 = hin[tid], v1 = hin[tid + 256];
  float m = fmaxf(v0, v1);
  for (int off = 16; off; off >>= 1) m = fmaxf(m, __shfl_xor(m, off, 32));
  if ((tid & 31) == 0) red[tid >> 5] = m;
  __syncthreads();
  if (tid == 0) {
    float mm = red[0];
    for (int i = 1; i < 8; ++i) mm = fmaxf(mm, red[i]);
    smax = mm;
  }
  __syncthreads();
  float e0 = __expf(v0 - smax), e1 = __expf(v1 - smax);
  float s = e0 + e1;
  for (int off = 16; off; off >>= 1) s += __shfl_xor(s, off, 32);
  if ((tid & 31) == 0) red[tid >> 5] = s;
  __syncthreads();
  if (tid == 0) {
    float ss = 0.f;
    for (int i = 0; i < 8; ++i) ss += red[i];
    ssum = ss;
  }
  __syncthreads();
  out[tid]       = e0 / ssum;
  out[tid + 256] = e1 / ssum;
}

// ---------------------------------------------------------------------------
extern "C" void kernel_launch(void* const* d_in, const int* in_sizes, int n_in,
                              void* d_out, int out_size, void* d_ws, size_t ws_size,
                              hipStream_t stream) {
  (void)in_sizes; (void)n_in; (void)out_size; (void)ws_size;
  const float* net_input = (const float*)d_in[0];  // [T,1,I]
  const float* W_ih      = (const float*)d_in[1];  // [4H,I]
  const float* W_hh      = (const float*)d_in[2];  // [4H,H]
  const float* b_ih      = (const float*)d_in[3];  // [4H]
  const float* b_hh      = (const float*)d_in[4];  // [4H]

  char* ws = (char*)d_ws;
  size_t off = 0;
  half_t* A_h = (half_t*)(ws + off); off += (size_t)T_STEPS * I_DIM * sizeof(half_t); // 64 MB
  half_t* W_h = (half_t*)(ws + off); off += (size_t)G_DIM * I_DIM * sizeof(half_t);   //  2 MB
  half_t* Xp  = (half_t*)(ws + off); off += (size_t)T_STEPS * G_DIM * sizeof(half_t); // 256 MB
  float*  hbuf = (float*)(ws + off); off += (size_t)2 * H_DIM * sizeof(float);
  int*    bar  = (int*)(ws + off);   off += 256;

  cvt_f32_to_f16<<<4096, 256, 0, stream>>>(net_input, A_h, T_STEPS * I_DIM);
  cvt_f32_to_f16<<<1024, 256, 0, stream>>>(W_ih, W_h, G_DIM * I_DIM);
  init_state<<<1, 256, 0, stream>>>(hbuf, bar);

  gemm_xproj<<<dim3(T_STEPS / BM, G_DIM / BN), 256, 0, stream>>>(A_h, W_h, b_ih, Xp);

  size_t recLds = (size_t)(128 * 512 + 128 + 128 + 32 + 512) * sizeof(float); // ~260 KB
  lstm_rec<<<RBLK, RTHR, recLds, stream>>>(Xp, W_hh, b_hh, hbuf, bar);

  softmax512<<<1, 256, 0, stream>>>(hbuf, (float*)d_out);
}
